// SelfAttn_29394756174437
// MI455X (gfx1250) — matrix operand order repaired and checked by reading, not compiled, not run on hardware
//
#include <hip/hip_runtime.h>
#include <hip/hip_bf16.h>

typedef __attribute__((ext_vector_type(16))) __bf16 v16bf;
typedef __attribute__((ext_vector_type(8)))  __bf16 v8bf;
typedef __attribute__((ext_vector_type(8)))  float  v8f;
typedef __attribute__((__vector_size__(16))) int    v4i_t;

#define BATCH 4
#define CCH   256
#define CKH   128
#define NPIX  4096
#define MQ    32
#define KC    32
#define BNELEM 16384.0f   // B*N
#define EPS   1e-5f

// softmax scale folded with log2(e):  exp(a/sqrt(128)) = exp2(a * C2)
#define C2 (0.08838834764831845f * 1.4426950408889634f)

// ---- CDNA5 async global->LDS copy (ASYNCcnt path), with safe fallback ----
#if defined(__AMDGCN__) && __has_builtin(__builtin_amdgcn_global_load_async_to_lds_b128)
#define HAS_ASYNC_LDS 1
#endif

__device__ __forceinline__ void g2l_b128(const __bf16* g, __bf16* l) {
#ifdef HAS_ASYNC_LDS
    __builtin_amdgcn_global_load_async_to_lds_b128(
        (__attribute__((address_space(1))) v4i_t*)(void*)g,
        (__attribute__((address_space(3))) v4i_t*)(void*)l, 0, 0);
#else
    *(v8bf*)l = *(const v8bf*)g;
#endif
}

__device__ __forceinline__ void wait_async_copies() {
#ifdef HAS_ASYNC_LDS
#if __has_builtin(__builtin_amdgcn_s_wait_asynccnt)
    __builtin_amdgcn_s_wait_asynccnt(0);
#else
    asm volatile("s_wait_asynccnt 0" ::: "memory");
#endif
#endif
}

// build a v16bf WMMA operand from two contiguous 16-byte runs
__device__ __forceinline__ v16bf ld16(const __bf16* p0, const __bf16* p1) {
    v8bf lo = *(const v8bf*)p0;
    v8bf hi = *(const v8bf*)p1;
    v16bf r;
#pragma unroll
    for (int j = 0; j < 8; ++j) { r[j] = lo[j]; r[j + 8] = hi[j]; }
    return r;
}

// ---------------- kernel 1: Q/K projection (1x1 conv) -> bf16, TRANSPOSED [b][n][ck] ----------------
__global__ __launch_bounds__(256) void proj_qk(const float* __restrict__ x,
                                               const float* __restrict__ wq,
                                               const float* __restrict__ bq,
                                               const float* __restrict__ wk,
                                               const float* __restrict__ bk,
                                               __bf16* __restrict__ Qt,
                                               __bf16* __restrict__ Kt) {
    const int n = blockIdx.x * 256 + threadIdx.x;
    const int o = blockIdx.y;
    const int b = blockIdx.z;
    const float* xb  = x + (size_t)b * CCH * NPIX + n;
    const float* wqr = wq + (size_t)o * CCH;
    const float* wkr = wk + (size_t)o * CCH;
    float aq = bq[o], ak = bk[o];
#pragma unroll 4
    for (int c = 0; c < CCH; ++c) {
        float xv = xb[(size_t)c * NPIX];
        aq = fmaf(wqr[c], xv, aq);
        ak = fmaf(wkr[c], xv, ak);
    }
    Qt[((size_t)b * NPIX + n) * CKH + o] = (__bf16)aq;
    Kt[((size_t)b * NPIX + n) * CKH + o] = (__bf16)ak;
}

// ---------------- kernel 2: fp32 -> bf16 convert ----------------
__global__ __launch_bounds__(256) void cvt_bf16(const float* __restrict__ src,
                                                __bf16* __restrict__ dst, int n) {
    int i = blockIdx.x * 256 + threadIdx.x;
    if (i < n) dst[i] = (__bf16)src[i];
}

// ---------------- kernel 3: flash attention ----------------
// grid (N/MQ, B), 256 threads (8 waves). Output OattT[b][n][c] bf16 (pre-wv), transposed.
__global__ __launch_bounds__(256) void flash_attn(const __bf16* __restrict__ Qt,
                                                  const __bf16* __restrict__ Kt,
                                                  const __bf16* __restrict__ Vbf,
                                                  __bf16* __restrict__ OattT) {
    __shared__ __align__(16) __bf16 k_lds[KC * CKH];     // [key][ch]   (8 KB, contiguous in global)
    __shared__ __align__(16) __bf16 v_lds[CCH * KC];     // [ch][key]   (16 KB)
    __shared__ __align__(16) __bf16 p_lds[MQ * KC];      // [query][key]
    __shared__ float row_max_part[2][MQ];
    __shared__ float row_sum_part[2][MQ];
    __shared__ float Mrun[MQ], Lrun[MQ], Rfac[MQ];

    const int tid  = threadIdx.x;
    const int wave = tid >> 5;
    const int lane = tid & 31;
    const int half = lane >> 4;
    const int ln   = lane & 15;
    const int b    = blockIdx.y;
    const int q0   = blockIdx.x * MQ;

    if (tid < MQ) { Mrun[tid] = -__builtin_inff(); Lrun[tid] = 0.0f; }

    v8f acc[4];
#pragma unroll
    for (int t = 0; t < 4; ++t)
#pragma unroll
        for (int g = 0; g < 8; ++g) acc[t][g] = 0.0f;

    const int m_sub = (wave >> 1) & 1;
    const int k_sub = wave & 1;
    v16bf q_a[4];
    if (wave < 4) {
        // A operand (16 queries x 32 ch per chunk): contiguous runs in Qt[n][ch]
        const __bf16* qrow = Qt + ((size_t)b * NPIX + q0 + m_sub * 16 + ln) * CKH;
#pragma unroll
        for (int i = 0; i < 4; ++i)
            q_a[i] = ld16(qrow + i * 32 + half * 8, qrow + i * 32 + 16 + half * 8);
    }
    __syncthreads();

    for (int n0 = 0; n0 < NPIX; n0 += KC) {
        // (a) stage K (8KB contiguous) and V tiles into LDS via async b128 copies
        {
            const __bf16* ksrc = Kt + ((size_t)b * NPIX + n0) * CKH;
            for (int i = tid; i < (KC * CKH) / 8; i += 256)        // 512 x 16B
                g2l_b128(ksrc + (size_t)i * 8, k_lds + i * 8);
            const __bf16* vsrc = Vbf + (size_t)b * CCH * NPIX;
            for (int i = tid; i < (CCH * KC) / 8; i += 256) {      // 1024 x 16B
                int c = i >> 2, part = i & 3;
                g2l_b128(vsrc + (size_t)c * NPIX + n0 + part * 8, v_lds + i * 8);
            }
            wait_async_copies();
        }
        __syncthreads();

        // (b) S = Q^T K  (16 queries x 16 keys tiles, K-depth 128 channels)
        v8f s;
#pragma unroll
        for (int g = 0; g < 8; ++g) s[g] = 0.0f;
        if (wave < 4) {
            const __bf16* kcol = k_lds + (k_sub * 16 + ln) * CKH;
#pragma unroll
            for (int i = 0; i < 4; ++i) {
                v16bf bb = ld16(kcol + i * 32 + half * 8, kcol + i * 32 + 16 + half * 8);
                s = __builtin_amdgcn_wmma_f32_16x16x32_bf16(false, q_a[i], false, bb,
                                                            (short)0, s, false, false);
            }
            // per-row (query) max across the 16 key lanes
#pragma unroll
            for (int g = 0; g < 8; ++g) {
                float v = s[g];
                v = fmaxf(v, __shfl_xor(v, 1, 32));
                v = fmaxf(v, __shfl_xor(v, 2, 32));
                v = fmaxf(v, __shfl_xor(v, 4, 32));
                v = fmaxf(v, __shfl_xor(v, 8, 32));
                if (ln == 0) row_max_part[k_sub][m_sub * 16 + half * 8 + g] = v;
            }
        }
        __syncthreads();

        // (c) online-softmax bookkeeping
        if (tid < MQ) {
            float mo = Mrun[tid];
            float mn = fmaxf(mo, fmaxf(row_max_part[0][tid], row_max_part[1][tid]));
            float r  = exp2f((mo - mn) * C2);
            Rfac[tid] = r; Mrun[tid] = mn; Lrun[tid] *= r;
        }
        __syncthreads();

        // (d) rescale accumulators; compute P = exp2((S - M)*C2), stage P (row-major [query][key])
#pragma unroll
        for (int t = 0; t < 4; ++t) {
            float rr = Rfac[(t & 1) * 16 + ln];
#pragma unroll
            for (int g = 0; g < 8; ++g) acc[t][g] *= rr;
        }
        if (wave < 4) {
#pragma unroll
            for (int g = 0; g < 8; ++g) {
                int m = m_sub * 16 + half * 8 + g;
                float p = exp2f((s[g] - Mrun[m]) * C2);
                p_lds[m * KC + k_sub * 16 + ln] = (__bf16)p;
                float v = p;
                v += __shfl_xor(v, 1, 32);
                v += __shfl_xor(v, 2, 32);
                v += __shfl_xor(v, 4, 32);
                v += __shfl_xor(v, 8, 32);
                if (ln == 0) row_sum_part[k_sub][m] = v;
            }
        }
        __syncthreads();

        if (tid < MQ) Lrun[tid] += row_sum_part[0][tid] + row_sum_part[1][tid];

        // (e) O += V * P^T   (wave w owns channel rows 32w..32w+31, both query subtiles)
#pragma unroll
        for (int t = 0; t < 4; ++t) {
            int c_tile = wave * 2 + (t >> 1);
            int ms = t & 1;
            const __bf16* vrow = v_lds + (c_tile * 16 + ln) * KC;
            const __bf16* prow = p_lds + (ms * 16 + ln) * KC + half * 16;
            v16bf av = ld16(vrow + half * 16, vrow + half * 16 + 8);
            v16bf bb = ld16(prow, prow + 8);
            acc[t] = __builtin_amdgcn_wmma_f32_16x16x32_bf16(false, av, false, bb,
                                                             (short)0, acc[t], false, false);
        }
        __syncthreads();
    }

    // epilogue: normalize by softmax denominator, write transposed bf16 [b][n][c] (packed 16B stores)
#pragma unroll
    for (int t = 0; t < 4; ++t) {
        int c_tile = wave * 2 + (t >> 1);
        int ms = t & 1;
        float invl = 1.0f / Lrun[ms * 16 + ln];
        v8bf vo;
#pragma unroll
        for (int g = 0; g < 8; ++g) vo[g] = (__bf16)(acc[t][g] * invl);
        size_t row = (size_t)b * NPIX + q0 + ms * 16 + ln;
        *(v8bf*)&OattT[row * CCH + c_tile * 16 + half * 8] = vo;
    }
}

// ---------------- kernel 4: out2 = wv * Oatt + bv  (WMMA) + BN partial stats ----------------
// one 16x16 output tile per wave; grid = (16 o_tiles * B*N/16 pixel tiles) / 8 waves
__global__ __launch_bounds__(256) void gemm_wv(const __bf16* __restrict__ WVbf,
                                               const __bf16* __restrict__ OattT,
                                               const float* __restrict__ bv,
                                               float* __restrict__ out2,
                                               float* __restrict__ stats) {
    const int tid  = threadIdx.x;
    const int wave = tid >> 5;
    const int lane = tid & 31;
    const int half = lane >> 4;
    const int ln   = lane & 15;
    const int gw   = blockIdx.x * 8 + wave;
    const int o_tile = gw & 15;
    const int ptile  = gw >> 4;               // 0..1023
    const int b      = ptile >> 8;            // 256 pixel-tiles per batch
    const int ncol0  = (ptile & 255) * 16;

    v8f acc;
#pragma unroll
    for (int g = 0; g < 8; ++g) acc[g] = 0.0f;

    const __bf16* arow = WVbf + (size_t)(o_tile * 16 + ln) * CCH;
    const __bf16* brow = OattT + ((size_t)b * NPIX + ncol0 + ln) * CCH;
#pragma unroll
    for (int i = 0; i < 8; ++i) {             // K over 256 channels
        v16bf av = ld16(arow + i * 32 + half * 8, arow + i * 32 + 16 + half * 8);
        v16bf bb = ld16(brow + i * 32 + half * 8, brow + i * 32 + 16 + half * 8);
        acc = __builtin_amdgcn_wmma_f32_16x16x32_bf16(false, av, false, bb,
                                                      (short)0, acc, false, false);
    }

    float* sums  = stats;
    float* sumsq = stats + CCH;
#pragma unroll
    for (int g = 0; g < 8; ++g) {
        int c = o_tile * 16 + half * 8 + g;
        float val = acc[g] + bv[c];
        out2[(size_t)b * CCH * NPIX + (size_t)c * NPIX + ncol0 + ln] = val;
        float sv = val, qv = val * val;
        sv += __shfl_xor(sv, 1, 32); qv += __shfl_xor(qv, 1, 32);
        sv += __shfl_xor(sv, 2, 32); qv += __shfl_xor(qv, 2, 32);
        sv += __shfl_xor(sv, 4, 32); qv += __shfl_xor(qv, 4, 32);
        sv += __shfl_xor(sv, 8, 32); qv += __shfl_xor(qv, 8, 32);
        if (ln == 0) {
            atomicAdd(&sums[c], sv);
            atomicAdd(&sumsq[c], qv);
        }
    }
}

// ---------------- kernel 5: batchnorm (training stats) + residual ----------------
__global__ __launch_bounds__(256) void bn_apply(const float* __restrict__ out2,
                                                const float* __restrict__ x,
                                                const float* __restrict__ gamma,
                                                const float* __restrict__ beta,
                                                const float* __restrict__ stats,
                                                float* __restrict__ out) {
    size_t idx = (size_t)blockIdx.x * 256 + threadIdx.x;
    int c = (int)((idx >> 12) & 255);         // N = 4096 = 2^12
    float mean = stats[c] * (1.0f / BNELEM);
    float var  = stats[CCH + c] * (1.0f / BNELEM) - mean * mean;
    float y = (out2[idx] - mean) * rsqrtf(var + EPS) * gamma[c] + beta[c] + x[idx];
    out[idx] = y;
}

// ---------------- launch ----------------
extern "C" void kernel_launch(void* const* d_in, const int* in_sizes, int n_in,
                              void* d_out, int out_size, void* d_ws, size_t ws_size,
                              hipStream_t stream) {
    const float* x     = (const float*)d_in[0];
    const float* wq    = (const float*)d_in[1];
    const float* bq    = (const float*)d_in[2];
    const float* wk    = (const float*)d_in[3];
    const float* bk    = (const float*)d_in[4];
    const float* wv    = (const float*)d_in[5];
    const float* bv    = (const float*)d_in[6];
    const float* gamma = (const float*)d_in[7];
    const float* beta  = (const float*)d_in[8];

    char* ws = (char*)d_ws;
    const size_t QT_OFF    = 0;                                           // 4 MB
    const size_t KT_OFF    = QT_OFF   + (size_t)BATCH * CKH * NPIX * 2;   // 4 MB
    const size_t VBF_OFF   = KT_OFF   + (size_t)BATCH * CKH * NPIX * 2;   // 8 MB
    const size_t WVBF_OFF  = VBF_OFF  + (size_t)BATCH * CCH * NPIX * 2;   // 128 KB
    const size_t OATT_OFF  = WVBF_OFF + (size_t)CCH * CCH * 2;            // 8 MB
    const size_t OUT2_OFF  = OATT_OFF + (size_t)BATCH * CCH * NPIX * 2;   // 16 MB
    const size_t STATS_OFF = OUT2_OFF + (size_t)BATCH * CCH * NPIX * 4;   // 2 KB

    __bf16* Qt    = (__bf16*)(ws + QT_OFF);
    __bf16* Kt    = (__bf16*)(ws + KT_OFF);
    __bf16* Vbf   = (__bf16*)(ws + VBF_OFF);
    __bf16* WVbf  = (__bf16*)(ws + WVBF_OFF);
    __bf16* OattT = (__bf16*)(ws + OATT_OFF);
    float*  out2  = (float*)(ws + OUT2_OFF);
    float*  stats = (float*)(ws + STATS_OFF);

    // 1) Q/K projection (writes [b][n][ck] transposed)
    proj_qk<<<dim3(NPIX / 256, CKH, BATCH), 256, 0, stream>>>(x, wq, bq, wk, bk, Qt, Kt);
    // 2) bf16 conversions for V (= x, [b][c][n]) and wv
    cvt_bf16<<<(BATCH * CCH * NPIX) / 256, 256, 0, stream>>>(x, Vbf, BATCH * CCH * NPIX);
    cvt_bf16<<<(CCH * CCH) / 256, 256, 0, stream>>>(wv, WVbf, CCH * CCH);
    // 3) flash attention (writes OattT [b][n][c])
    flash_attn<<<dim3(NPIX / MQ, BATCH), 256, 0, stream>>>(Qt, Kt, Vbf, OattT);
    // 4) wv GEMM + BN partial sums
    (void)hipMemsetAsync(stats, 0, 2 * CCH * sizeof(float), stream);
    gemm_wv<<<(16 * (BATCH * NPIX / 16)) / 8, 256, 0, stream>>>(WVbf, OattT, bv, out2, stats);
    // 5) batchnorm + residual
    bn_apply<<<(BATCH * CCH * NPIX) / 256, 256, 0, stream>>>(out2, x, gamma, beta, stats,
                                                             (float*)d_out);
}
